// Summarizer_16329465660227
// MI455X (gfx1250) — compile-verified, matrix operand order checked by reading, not run on hardware
//
#include <hip/hip_runtime.h>
#include <hip/hip_bf16.h>
#include <math.h>

typedef __attribute__((ext_vector_type(16))) _Float16 v16h;
typedef __attribute__((ext_vector_type(8)))  float    v8f;

// Problem constants (match reference setup_inputs)
#define BB   64
#define EE   8
#define NN   32768
#define DD   128
#define RR   16
#define STEPC 256

#define NPD  (NN + 64)        // padded dry_h pitch (zeros beyond N)
#define XC   (NN + 256)       // kernel reversal center: kernR[x] = kern[XC - x]
#define LR   (NN + 2560)      // padded reversed-kernel pitch (35328 = 138*256)

union U16 {
    v16h     v;
    uint32_t u[8];
    _Float16 h[16];
};

// ---------------------------------------------------------------------------
// prep: room_w = softmax(g @ w_room + b_room); mix = sigmoid(g @ w_mix + b_mix)
// ---------------------------------------------------------------------------
__global__ __launch_bounds__(64) void prep_kernel(
    const float* __restrict__ g, const float* __restrict__ w_room,
    const float* __restrict__ b_room, const float* __restrict__ w_mix,
    const float* __restrict__ b_mix, float* __restrict__ room_w,
    float* __restrict__ mixv)
{
    int b = threadIdx.x;
    if (b >= BB) return;
    const float* gr = g + b * DD;
    float logit[RR];
    float mx = -3.4e38f;
    for (int r = 0; r < RR; ++r) {
        float s = b_room[r];
        for (int d = 0; d < DD; ++d) s = fmaf(gr[d], w_room[d * RR + r], s);
        logit[r] = s;
        mx = fmaxf(mx, s);
    }
    float den = 0.f;
    for (int r = 0; r < RR; ++r) { logit[r] = expf(logit[r] - mx); den += logit[r]; }
    float inv = 1.0f / den;
    for (int r = 0; r < RR; ++r) room_w[b * RR + r] = logit[r] * inv;

    float m = b_mix[0];
    for (int d = 0; d < DD; ++d) m = fmaf(gr[d], w_mix[d], m);
    mixv[b] = 1.0f / (1.0f + expf(-m));
}

// ---------------------------------------------------------------------------
// build_dry: dry[b,t] = sum_e events[b,e,t-start]  (gather form of scatter-add)
// ---------------------------------------------------------------------------
__global__ __launch_bounds__(256) void build_dry_kernel(
    const float* __restrict__ events, const int* __restrict__ indices,
    float* __restrict__ dry, _Float16* __restrict__ dryh)
{
    int b = blockIdx.y;
    int t = blockIdx.x * 256 + threadIdx.x;
    if (t >= NPD) return;
    if (t >= NN) { dryh[(size_t)b * NPD + t] = (_Float16)0.0f; return; }
    float acc = 0.f;
#pragma unroll
    for (int e = 0; e < EE; ++e) {
        int start = indices[b * EE + e] * STEPC;
        int i = t - start;                       // i < N automatic (t < N)
        if (i >= 0) acc += events[((size_t)(b * EE + e)) * NN + i];
    }
    dry[(size_t)b * NN + t] = acc;
    dryh[(size_t)b * NPD + t] = (_Float16)acc;
}

// ---------------------------------------------------------------------------
// build_kern: kernR[b,x] = (f16) sum_r room_w[b,r]*rooms[r, XC - x], zero-padded
// (reversed layout makes the WMMA A-fragment contiguous ascending in memory)
// ---------------------------------------------------------------------------
__global__ __launch_bounds__(256) void build_kern_kernel(
    const float* __restrict__ rooms, const float* __restrict__ room_w,
    _Float16* __restrict__ kernR)
{
    int b = blockIdx.y;
    int x = blockIdx.x * 256 + threadIdx.x;      // LR == 138*256, exact
    int y = XC - x;
    float v = 0.f;
    if (y >= 0 && y < NN) {
        const float* w = room_w + b * RR;
#pragma unroll
        for (int r = 0; r < RR; ++r) v = fmaf(w[r], rooms[(size_t)r * NN + y], v);
    }
    kernR[(size_t)b * LR + x] = (_Float16)v;
}

// ---------------------------------------------------------------------------
// conv: causal time-domain convolution as Toeplitz x Hankel WMMA GEMM.
// Grid (B, N/2048), block 128 (4 waves). Wave w owns 512 samples = two 16x16
// f32 accumulator tiles; the dry (B) fragment is shared by both tiles.
//   D[i,j] += A[i,k] * B[k,j]
//   A[i,k] = kern[t0+16i - tau - k] = kernR[XC - t0 - 16i + tau + k]
//   B[k,j] = dry[tau + k + j]
// 16-bit fragment layout (ISA 7.12.2): lane L: row/col = L&15, c = 8*(L>=16),
// halves h<8 -> k = c+h, h>=8 -> k = c+8+h.
// ---------------------------------------------------------------------------
__global__ __launch_bounds__(128) void conv_kernel(
    const _Float16* __restrict__ dryh, const _Float16* __restrict__ kernR,
    float* __restrict__ wet)
{
    const int b    = blockIdx.x;
    const int tblk = blockIdx.y * 2048;
    const int wave = threadIdx.x >> 5;
    const int lane = threadIdx.x & 31;
    const int col  = lane & 15;            // j (B/D column) == i (A row)
    const int hi   = lane >> 4;            // lane group
    const int c    = hi * 8;               // K offset of this lane group

    const int t0a = tblk + wave * 512;
    const int t0b = t0a + 256;

    const _Float16* dh = dryh  + (size_t)b * NPD;
    const _Float16* kr = kernR + (size_t)b * LR;

    v8f acc0 = {};
    v8f acc1 = {};

    const int tauEnd = tblk + 2048;        // causal bound for this block
    for (int tau = 0; tau < tauEnd; tau += 32) {
        // ---- B fragment (dry Hankel), shared by both output tiles ----
        U16 bf;
        const int b0 = tau + c + col;
#pragma unroll
        for (int h = 0; h < 8; ++h)  bf.h[h] = dh[b0 + h];
#pragma unroll
        for (int h = 8; h < 16; ++h) bf.h[h] = dh[b0 + h + 8];

        // ---- A fragments (reversed kernel Toeplitz): contiguous dwords ----
        const int a0a = XC - t0a - 16 * col + tau + c;   // always even
        const uint32_t* pa = (const uint32_t*)(kr + a0a);
        const uint32_t* pb = (const uint32_t*)(kr + (a0a - 256));
        U16 A0, A1;
#pragma unroll
        for (int t = 0; t < 4; ++t) {
            A0.u[t] = pa[t];  A0.u[t + 4] = pa[t + 8];
            A1.u[t] = pb[t];  A1.u[t + 4] = pb[t + 8];
        }
        __builtin_prefetch(kr + a0a + 64, 0, 1);   // next slice -> global_prefetch

        acc0 = __builtin_amdgcn_wmma_f32_16x16x32_f16(
            false, A0.v, false, bf.v, (short)0, acc0, false, false);
        acc1 = __builtin_amdgcn_wmma_f32_16x16x32_f16(
            false, A1.v, false, bf.v, (short)0, acc1, false, false);
    }

    // C/D layout: VGPR r, lane L -> row i = r + 8*hi, col j = L&15
    float* wrow = wet + (size_t)b * NN;
#pragma unroll
    for (int r = 0; r < 8; ++r) {
        int i = r + hi * 8;
        wrow[t0a + 16 * i + col] = acc0[r];
        wrow[t0b + 16 * i + col] = acc1[r];
    }
}

// ---------------------------------------------------------------------------
// finalize: out = mix*wet + (1-mix)*dry, then divide by (rowmax + 1e-8)
// ---------------------------------------------------------------------------
__global__ __launch_bounds__(256) void finalize_kernel(
    const float* __restrict__ wet, const float* __restrict__ dry,
    const float* __restrict__ mixv, float* __restrict__ out)
{
    __shared__ float red[256];
    const int b   = blockIdx.x;
    const int tid = threadIdx.x;
    const float mix = mixv[b];
    const float omx = 1.0f - mix;
    const size_t base = (size_t)b * NN;

    float lm = -3.4e38f;
    for (int t = tid; t < NN; t += 256) {
        float v = mix * wet[base + t] + omx * dry[base + t];
        out[base + t] = v;
        lm = fmaxf(lm, v);
    }
    red[tid] = lm;
    __syncthreads();
    for (int s = 128; s > 0; s >>= 1) {
        if (tid < s) red[tid] = fmaxf(red[tid], red[tid + s]);
        __syncthreads();
    }
    const float scale = 1.0f / (red[0] + 1e-8f);
    for (int t = tid; t < NN; t += 256) out[base + t] *= scale;
}

// ---------------------------------------------------------------------------
extern "C" void kernel_launch(void* const* d_in, const int* in_sizes, int n_in,
                              void* d_out, int out_size, void* d_ws, size_t ws_size,
                              hipStream_t stream)
{
    const float* events  = (const float*)d_in[0];
    const int*   indices = (const int*)  d_in[1];
    const float* g       = (const float*)d_in[2];
    const float* rooms   = (const float*)d_in[3];
    const float* w_room  = (const float*)d_in[4];
    const float* b_room  = (const float*)d_in[5];
    const float* w_mix   = (const float*)d_in[6];
    const float* b_mix   = (const float*)d_in[7];
    float* out = (float*)d_out;

    char* ws = (char*)d_ws;
    size_t o_dry  = 0;                                 // B*N*4   = 8,388,608
    size_t o_wet  = o_dry  + (size_t)BB * NN * 4;      // B*N*4   = 8,388,608
    size_t o_dryh = o_wet  + (size_t)BB * NN * 4;      // B*NPD*2 = 4,202,496
    size_t o_kern = o_dryh + (size_t)BB * NPD * 2;     // B*LR*2  = 4,521,984
    size_t o_rw   = o_kern + (size_t)BB * LR * 2;      // 4 KB
    size_t o_mix  = o_rw   + (size_t)BB * RR * 4;      // 256 B

    float*    dry    = (float*)   (ws + o_dry);
    float*    wet    = (float*)   (ws + o_wet);
    _Float16* dryh   = (_Float16*)(ws + o_dryh);
    _Float16* kernR  = (_Float16*)(ws + o_kern);
    float*    room_w = (float*)   (ws + o_rw);
    float*    mixv   = (float*)   (ws + o_mix);

    prep_kernel<<<1, 64, 0, stream>>>(g, w_room, b_room, w_mix, b_mix, room_w, mixv);

    build_dry_kernel<<<dim3((NPD + 255) / 256, BB), 256, 0, stream>>>(
        events, indices, dry, dryh);

    build_kern_kernel<<<dim3(LR / 256, BB), 256, 0, stream>>>(rooms, room_w, kernR);

    conv_kernel<<<dim3(BB, NN / 2048), 128, 0, stream>>>(dryh, kernR, wet);

    finalize_kernel<<<BB, 256, 0, stream>>>(wet, dry, mixv, out);
}
